// DynamicCorrelationNet_84207128805732
// MI455X (gfx1250) — matrix operand order repaired and checked by reading, not compiled, not run on hardware
//
#include <hip/hip_runtime.h>
#include <hip/hip_bf16.h>

#define NPTS  8192
#define FIN   31
#define HDIM  256
#define KNN_K 10
#define NLAYER 3
#define KNN_WAVES 4
#define PANEL_ROWS 32   // columns of the distance matrix DMA'd per TDM issue

typedef __attribute__((ext_vector_type(16))) _Float16 v16h;
typedef __attribute__((ext_vector_type(8)))  _Float16 v8h;
typedef __attribute__((ext_vector_type(8)))  float    v8f;
typedef __attribute__((ext_vector_type(4)))  unsigned int v4u;
typedef __attribute__((ext_vector_type(8)))  int v8i;
typedef __attribute__((ext_vector_type(4)))  int v4i;

// ---------------------------------------------------------------------------
// TDM: DMA one PANEL_ROWS x 512-byte panel of hf16 (2-D tile) into LDS.
// D# packing per CDNA5 ISA:
//   group0: count=1 | lds_addr | global_addr[56:0] | type=2
//   group1: data_size=3 (8B units), tensor_dim0=64, tensor_dim1=8192,
//           tile_dim0=64, tile_dim1=PANEL_ROWS, tensor_dim0_stride=64
// Tracked with TENSORcnt; issued by one wave, completion published via barrier.
// ---------------------------------------------------------------------------
__device__ __forceinline__ void tdm_load_panel(const _Float16* gsrc, unsigned int lds_off) {
    const unsigned long long ga = (unsigned long long)(uintptr_t)gsrc;
    v4u g0;
    g0[0] = 1u;                                               // count=1 (valid user D#)
    g0[1] = lds_off;                                          // lds_addr (bytes)
    g0[2] = (unsigned int)ga;                                 // global_addr[31:0]
    g0[3] = (unsigned int)((ga >> 32) & 0x1FFFFFFu) | (2u << 30); // addr[56:32] | type=2
    v8i g1;
    g1[0] = (int)(3u << 16);        // workgroup_mask=0, data_size=3 (8 bytes)
    g1[1] = (int)(64u << 16);       // tensor_dim0[15:0]=64  (row length, 8B units)
    g1[2] = (int)(8192u << 16);     // tensor_dim1[15:0]=8192
    g1[3] = (int)(64u << 16);       // tile_dim0=64 (512B per row)
    g1[4] = PANEL_ROWS;             // tile_dim1=PANEL_ROWS rows, tile_dim2=0
    g1[5] = 64;                     // tensor_dim0_stride=64 units (=512B row pitch)
    g1[6] = 0;
    g1[7] = 0;
    v4i g2 = {0, 0, 0, 0};          // 2-D tile: groups 2/3 unused
    v4i g3 = {0, 0, 0, 0};
    v8i g4 = {0, 0, 0, 0, 0, 0, 0, 0};
    __builtin_amdgcn_tensor_load_to_lds(g0, g1, g2, g3, g4, 0);
}

// ---------------------------------------------------------------------------
// Fold BN into W1, split W1 into Wa = (top-bot)*scale, Wb = bot*scale, and
// store all WMMA B-operands transposed ([out][in], f16) so each lane's
// B-fragment load is 32 contiguous bytes.
// ---------------------------------------------------------------------------
__global__ void prep_weights_kernel(const float* __restrict__ W1, const float* __restrict__ b1,
                                    const float* __restrict__ gamma, const float* __restrict__ beta,
                                    const float* __restrict__ rmean, const float* __restrict__ rvar,
                                    const float* __restrict__ W2,
                                    _Float16* __restrict__ WaT, _Float16* __restrict__ WbT,
                                    _Float16* __restrict__ W2T, float* __restrict__ b1f) {
    const int l = blockIdx.x >> 8;        // layer
    const int o = blockIdx.x & 255;       // output channel
    const int i = threadIdx.x;            // input channel
    const float scale = gamma[l * HDIM + o] * rsqrtf(rvar[l * HDIM + o] + 1e-5f);
    const float wt = W1[((size_t)l * 2 * HDIM + i) * HDIM + o];
    const float wb = W1[((size_t)l * 2 * HDIM + HDIM + i) * HDIM + o];
    const size_t off = ((size_t)l * HDIM + o) * HDIM + i;
    WaT[off] = (_Float16)((wt - wb) * scale);
    WbT[off] = (_Float16)(wb * scale);
    W2T[off] = (_Float16)W2[((size_t)l * HDIM + i) * HDIM + o];
    if (i == 0)
        b1f[l * HDIM + o] = (b1[l * HDIM + o] - rmean[l * HDIM + o]) * scale + beta[l * HDIM + o];
}

// ---------------------------------------------------------------------------
// h0 = relu(x @ Wp + bp)   (8192 x 31 x 256, tiny -> plain VALU)
// ---------------------------------------------------------------------------
__global__ void proj_kernel(const float* __restrict__ x, const float* __restrict__ Wp,
                            const float* __restrict__ bp, float* __restrict__ h) {
    const int i = blockIdx.x, c = threadIdx.x;
    __shared__ float xs[FIN];
    if (c < FIN) xs[c] = x[i * FIN + c];
    __syncthreads();
    float acc = bp[c];
#pragma unroll
    for (int f = 0; f < FIN; ++f) acc = fmaf(xs[f], Wp[f * HDIM + c], acc);
    h[(size_t)i * HDIM + c] = acc > 0.f ? acc : 0.f;
}

// ---------------------------------------------------------------------------
// hf16 = (f16)h ; sq[i] = sum_c h[i][c]^2
// ---------------------------------------------------------------------------
__global__ void hprep_kernel(const float* __restrict__ h, _Float16* __restrict__ hf,
                             float* __restrict__ sq) {
    const int i = blockIdx.x, c = threadIdx.x;
    const float v = h[(size_t)i * HDIM + c];
    hf[(size_t)i * HDIM + c] = (_Float16)v;
    __shared__ float red[HDIM];
    red[c] = v * v;
    __syncthreads();
    for (int s = HDIM / 2; s > 0; s >>= 1) {
        if (c < s) red[c] += red[c + s];
        __syncthreads();
    }
    if (c == 0) sq[i] = red[0];
}

__device__ __forceinline__ void topk_insert(float (&kd)[KNN_K], int (&ki)[KNN_K], float d, int j) {
    if (d < kd[KNN_K - 1]) {
        kd[KNN_K - 1] = d; ki[KNN_K - 1] = j;
#pragma unroll
        for (int t = KNN_K - 2; t >= 0; --t) {
            if (kd[t + 1] < kd[t]) {
                float td = kd[t]; kd[t] = kd[t + 1]; kd[t + 1] = td;
                int   ti = ki[t]; ki[t] = ki[t + 1]; ki[t + 1] = ti;
            }
        }
    }
}

// ---------------------------------------------------------------------------
// KNN: 4 waves/block, each wave owns a 16-row tile; A (16x256 f16) lives in
// 64 VGPRs.  A PANEL_ROWS-col x 256-K f16 column panel (16 KB) is shared by
// all 4 waves through a double-buffered LDS panel filled by the Tensor Data
// Mover: wave 0 issues tensor_load_to_lds for panel t+1 while everyone
// computes panel t (two 16-col sub-tiles: clause of 16x ds_load_b128 ->
// 8x back-to-back v_wmma_f32_16x16x32_f16, pinned by a sched_barrier), then
// s_wait_tensorcnt 0 + workgroup barrier.  d2 = si+sj-2*dot, per-row top-10
// kept as a register insertion list (2 lanes per row, merged at the end).
// ---------------------------------------------------------------------------
__global__ void __launch_bounds__(32 * KNN_WAVES) knn_kernel(const _Float16* __restrict__ hf,
                                                             const float* __restrict__ sq,
                                                             int* __restrict__ knn) {
    const int tid = threadIdx.x;
    const int w = tid >> 5;
    const int lane = tid & 31;
    const int rowbase = blockIdx.x * (16 * KNN_WAVES) + w * 16;
    const int m = lane & 15;          // row within tile / column within panel
    const int kh = lane >> 4;         // K-half selector

    __shared__ __align__(32) _Float16 panel[2][PANEL_ROWS * HDIM];  // 2 x 16 KB
    __shared__ float ldsD[KNN_WAVES][16][16];
    __shared__ float ldsMd[KNN_WAVES][32][KNN_K];
    __shared__ int   ldsMi[KNN_WAVES][32][KNN_K];

    // A fragment: row m, K-chunk c covers K = c*32 .. c*32+31
    v16h A[8];
    const _Float16* arow = hf + (size_t)(rowbase + m) * HDIM;
#pragma unroll
    for (int c = 0; c < 8; ++c) {
        v8h lo = *(const v8h*)(arow + c * 32 + kh * 16);
        v8h hi = *(const v8h*)(arow + c * 32 + kh * 16 + 8);
        A[c] = __builtin_shufflevector(lo, hi, 0,1,2,3,4,5,6,7,8,9,10,11,12,13,14,15);
    }
    float si[8];
#pragma unroll
    for (int r = 0; r < 8; ++r) si[r] = sq[rowbase + kh * 8 + r];

    float kd[KNN_K]; int ki[KNN_K];
#pragma unroll
    for (int t = 0; t < KNN_K; ++t) { kd[t] = 3.4e38f; ki[t] = 0; }

    // prime panel 0
    if (tid < 32) {
        tdm_load_panel(hf, (unsigned int)(uintptr_t)&panel[0][0]);
        __builtin_amdgcn_s_wait_tensorcnt(0);
    }
    __syncthreads();

    for (int it = 0; it < NPTS / PANEL_ROWS; ++it) {
        const int cur = it & 1;
        const int cb = it * PANEL_ROWS;
        // kick off DMA of the next panel into the other buffer
        if ((it + 1 < NPTS / PANEL_ROWS) && tid < 32)
            tdm_load_panel(hf + (size_t)(cb + PANEL_ROWS) * HDIM,
                           (unsigned int)(uintptr_t)&panel[cur ^ 1][0]);

#pragma unroll
        for (int s = 0; s < PANEL_ROWS / 16; ++s) {
            const int colbase = cb + s * 16;
            // B fragments from LDS panel (column n = m, K-half kh)
            const _Float16* brow = &panel[cur][(s * 16 + m) * HDIM + kh * 16];
            v16h B[8];
#pragma unroll
            for (int c = 0; c < 8; ++c) B[c] = *(const v16h*)(brow + c * 32);
            __builtin_amdgcn_sched_barrier(0);   // clause loads; keep WMMAs after
            v8f acc = {};
#pragma unroll
            for (int c = 0; c < 8; ++c)
                acc = __builtin_amdgcn_wmma_f32_16x16x32_f16(false, A[c], false, B[c],
                                                             (short)0, acc, false, false);

            const float sj = sq[colbase + m];
            // C layout: VGPR r, lane -> row = r + 8*kh, col = m
#pragma unroll
            for (int r = 0; r < 8; ++r)
                ldsD[w][kh * 8 + r][m] = si[r] + sj - 2.0f * acc[r];
            asm volatile("s_wait_dscnt 0" ::: "memory");  // wave-local cross-lane exchange
#pragma unroll
            for (int c = 0; c < 8; ++c)
                topk_insert(kd, ki, ldsD[w][m][kh * 8 + c], colbase + kh * 8 + c);
        }

        if (tid < 32) __builtin_amdgcn_s_wait_tensorcnt(0);
        __syncthreads();   // next-panel DMA complete + everyone done with 'cur'
    }

    // merge the two half-row lists (lane, lane+16)
#pragma unroll
    for (int t = 0; t < KNN_K; ++t) { ldsMd[w][lane][t] = kd[t]; ldsMi[w][lane][t] = ki[t]; }
    asm volatile("s_wait_dscnt 0" ::: "memory");
    if (lane < 16) {
#pragma unroll
        for (int t = 0; t < KNN_K; ++t)
            topk_insert(kd, ki, ldsMd[w][lane + 16][t], ldsMi[w][lane + 16][t]);
#pragma unroll
        for (int t = 0; t < KNN_K; ++t)
            knn[(rowbase + lane) * KNN_K + t] = ki[t];
    }
}

// ---------------------------------------------------------------------------
// P = h @ Wa + b1'   and   Q = h @ Wb      (h in f16, WMMA, A reused for both;
// B fragments batched + sched_barrier so each WMMA chain runs back-to-back)
// ---------------------------------------------------------------------------
__global__ void __launch_bounds__(32) pq_gemm_kernel(const _Float16* __restrict__ hf,
                                                     const _Float16* __restrict__ WaT,
                                                     const _Float16* __restrict__ WbT,
                                                     const float* __restrict__ b1f,
                                                     float* __restrict__ P, float* __restrict__ Q) {
    const int lane = threadIdx.x & 31;
    const int rowbase = blockIdx.x * 16;
    const int n = lane & 15;
    const int kh = lane >> 4;

    v16h A[8];
    const _Float16* arow = hf + (size_t)(rowbase + n) * HDIM;
#pragma unroll
    for (int c = 0; c < 8; ++c) {
        v8h lo = *(const v8h*)(arow + c * 32 + kh * 16);
        v8h hi = *(const v8h*)(arow + c * 32 + kh * 16 + 8);
        A[c] = __builtin_shufflevector(lo, hi, 0,1,2,3,4,5,6,7,8,9,10,11,12,13,14,15);
    }

    for (int nt = 0; nt < 16; ++nt) {
        const _Float16* ba = WaT + (size_t)(nt * 16 + n) * HDIM + kh * 16;
        const _Float16* bb = WbT + (size_t)(nt * 16 + n) * HDIM + kh * 16;
        v16h B[8];
        v8f accp = {}, accq = {};
#pragma unroll
        for (int c = 0; c < 8; ++c) B[c] = *(const v16h*)(ba + c * 32);
        __builtin_amdgcn_sched_barrier(0);
#pragma unroll
        for (int c = 0; c < 8; ++c)
            accp = __builtin_amdgcn_wmma_f32_16x16x32_f16(false, A[c], false, B[c],
                                                          (short)0, accp, false, false);
#pragma unroll
        for (int c = 0; c < 8; ++c) B[c] = *(const v16h*)(bb + c * 32);
        __builtin_amdgcn_sched_barrier(0);
#pragma unroll
        for (int c = 0; c < 8; ++c)
            accq = __builtin_amdgcn_wmma_f32_16x16x32_f16(false, A[c], false, B[c],
                                                          (short)0, accq, false, false);
        const float bias = b1f[nt * 16 + n];
#pragma unroll
        for (int r = 0; r < 8; ++r) {
            const size_t off = (size_t)(rowbase + kh * 8 + r) * HDIM + nt * 16 + n;
            P[off] = accp[r] + bias;
            Q[off] = accq[r];
        }
    }
}

// ---------------------------------------------------------------------------
// Edge MLP + max-aggregation: one wave per node.  A rows = 10 edges
// m_k = relu(P_i + Q_{j_k}) (rows 10..15 zero-padded), C = A @ W2 (16 N-tiles),
// max over valid rows taken from the C-fragment layout + lane^16 shuffle.
// h_out = relu(max + b2) + h_in.
// ---------------------------------------------------------------------------
__global__ void __launch_bounds__(32) edge_kernel(const float* __restrict__ P,
                                                  const float* __restrict__ Q,
                                                  const int* __restrict__ knn,
                                                  const _Float16* __restrict__ W2T,
                                                  const float* __restrict__ b2l,
                                                  const float* __restrict__ h_in,
                                                  float* __restrict__ h_out) {
    const int lane = threadIdx.x & 31;
    const int i = blockIdx.x;
    const int m = lane & 15;   // edge row in A / output column in C
    const int kh = lane >> 4;
    const bool valid = (m < KNN_K);
    const int j = valid ? knn[i * KNN_K + m] : i;

    const float* prow = P + (size_t)i * HDIM;
    const float* qrow = Q + (size_t)j * HDIM;

    v16h A[8];
#pragma unroll
    for (int c = 0; c < 8; ++c) {
        v16h a;
#pragma unroll
        for (int g = 0; g < 2; ++g) {
            const int kb = c * 32 + kh * 16 + g * 8;
#pragma unroll
            for (int t = 0; t < 8; ++t) {
                float v = prow[kb + t] + qrow[kb + t];
                v = v > 0.f ? v : 0.f;
                a[g * 8 + t] = valid ? (_Float16)v : (_Float16)0.f;
            }
        }
        A[c] = a;
    }

#pragma unroll 1
    for (int nt = 0; nt < 16; ++nt) {
        const _Float16* bb = W2T + (size_t)(nt * 16 + m) * HDIM + kh * 16;
        v16h B[8];
#pragma unroll
        for (int c = 0; c < 8; ++c) B[c] = *(const v16h*)(bb + c * 32);
        __builtin_amdgcn_sched_barrier(0);
        v8f acc = {};
#pragma unroll
        for (int c = 0; c < 8; ++c)
            acc = __builtin_amdgcn_wmma_f32_16x16x32_f16(false, A[c], false, B[c],
                                                         (short)0, acc, false, false);
        // rows 0-7 live in lanes 0-15 (all valid), rows 8-9 in lanes 16-31
        float vmax;
        if (kh == 0) {
            vmax = acc[0];
#pragma unroll
            for (int r = 1; r < 8; ++r) vmax = fmaxf(vmax, acc[r]);
        } else {
            vmax = fmaxf(acc[0], acc[1]);   // rows 8,9 only (10..15 are padding)
        }
        const float other = __shfl_xor(vmax, 16, 32);
        if (lane < 16) {
            float fin = fmaxf(vmax, other) + b2l[nt * 16 + m];
            fin = fin > 0.f ? fin : 0.f;
            const size_t off = (size_t)i * HDIM + nt * 16 + m;
            h_out[off] = fin + h_in[off];
        }
    }
}

// ---------------------------------------------------------------------------
// out = relu(h @ Wo1 + bo1) @ Wo2 + bo2   (small head, VALU + LDS reduce)
// ---------------------------------------------------------------------------
__global__ void head_kernel(const float* __restrict__ h, const float* __restrict__ Wo1,
                            const float* __restrict__ bo1, const float* __restrict__ Wo2,
                            const float* __restrict__ bo2, float* __restrict__ out) {
    const int i = blockIdx.x, c = threadIdx.x;   // 128 threads
    __shared__ float hs[HDIM];
    hs[c] = h[(size_t)i * HDIM + c];
    hs[c + 128] = h[(size_t)i * HDIM + c + 128];
    __syncthreads();
    float t = bo1[c];
#pragma unroll 4
    for (int f = 0; f < HDIM; ++f) t = fmaf(hs[f], Wo1[f * 128 + c], t);
    t = t > 0.f ? t : 0.f;
    __shared__ float red[128];
    red[c] = t * Wo2[c];
    __syncthreads();
    for (int s = 64; s > 0; s >>= 1) {
        if (c < s) red[c] += red[c + s];
        __syncthreads();
    }
    if (c == 0) out[i] = red[0] + bo2[0];
}

// ---------------------------------------------------------------------------
extern "C" void kernel_launch(void* const* d_in, const int* in_sizes, int n_in,
                              void* d_out, int out_size, void* d_ws, size_t ws_size,
                              hipStream_t stream) {
    const float* x     = (const float*)d_in[0];
    /* d_in[1] = batch: all zeros -> single graph, same_graph is all-true */
    const float* Wp    = (const float*)d_in[2];
    const float* bp    = (const float*)d_in[3];
    const float* W1    = (const float*)d_in[4];
    const float* b1    = (const float*)d_in[5];
    const float* gamma = (const float*)d_in[6];
    const float* beta  = (const float*)d_in[7];
    const float* rmean = (const float*)d_in[8];
    const float* rvar  = (const float*)d_in[9];
    const float* W2    = (const float*)d_in[10];
    const float* b2    = (const float*)d_in[11];
    const float* Wo1   = (const float*)d_in[12];
    const float* bo1   = (const float*)d_in[13];
    const float* Wo2   = (const float*)d_in[14];
    const float* bo2   = (const float*)d_in[15];
    float* out = (float*)d_out;

    char* ws = (char*)d_ws;
    auto alloc = [&](size_t bytes) -> char* {
        char* p = ws;
        ws += (bytes + 255) & ~(size_t)255;
        return p;
    };
    float*    hA   = (float*)alloc((size_t)NPTS * HDIM * 4);
    float*    hB   = (float*)alloc((size_t)NPTS * HDIM * 4);
    _Float16* hf   = (_Float16*)alloc((size_t)NPTS * HDIM * 2);
    float*    sq   = (float*)alloc((size_t)NPTS * 4);
    int*      knn  = (int*)alloc((size_t)NPTS * KNN_K * 4);
    float*    P    = (float*)alloc((size_t)NPTS * HDIM * 4);
    float*    Q    = (float*)alloc((size_t)NPTS * HDIM * 4);
    _Float16* WaT  = (_Float16*)alloc((size_t)NLAYER * HDIM * HDIM * 2);
    _Float16* WbT  = (_Float16*)alloc((size_t)NLAYER * HDIM * HDIM * 2);
    _Float16* W2T  = (_Float16*)alloc((size_t)NLAYER * HDIM * HDIM * 2);
    float*    b1f  = (float*)alloc((size_t)NLAYER * HDIM * 4);

    prep_weights_kernel<<<NLAYER * HDIM, HDIM, 0, stream>>>(
        W1, b1, gamma, beta, rmean, rvar, W2, WaT, WbT, W2T, b1f);
    proj_kernel<<<NPTS, HDIM, 0, stream>>>(x, Wp, bp, hA);

    float* hc = hA;
    float* hn = hB;
    for (int l = 0; l < NLAYER; ++l) {
        hprep_kernel<<<NPTS, HDIM, 0, stream>>>(hc, hf, sq);
        knn_kernel<<<NPTS / (16 * KNN_WAVES), 32 * KNN_WAVES, 0, stream>>>(hf, sq, knn);
        pq_gemm_kernel<<<NPTS / 16, 32, 0, stream>>>(
            hf, WaT + (size_t)l * HDIM * HDIM, WbT + (size_t)l * HDIM * HDIM,
            b1f + l * HDIM, P, Q);
        edge_kernel<<<NPTS, 32, 0, stream>>>(
            P, Q, knn, W2T + (size_t)l * HDIM * HDIM, b2 + l * HDIM, hc, hn);
        float* t = hc; hc = hn; hn = t;
    }
    head_kernel<<<NPTS, 128, 0, stream>>>(hc, Wo1, bo1, Wo2, bo2, out);
}
